// Encoder_29085518528711
// MI455X (gfx1250) — matrix operand order repaired and checked by reading, not compiled, not run on hardware
//
#include <hip/hip_runtime.h>
#include <hip/hip_bf16.h>

typedef __attribute__((ext_vector_type(16))) __bf16 v16bf;
typedef __attribute__((ext_vector_type(8)))  __bf16 v8bf;
typedef __attribute__((ext_vector_type(8)))  float  v8f;
typedef int i32x4 __attribute__((vector_size(16)));   // matches builtin's V4i pointee

#define IN_C 512
#define HID  64   // width of both GEMM outputs (layer1: 64; layer2: [mu|logstd] = 32+32)

#define AS1 __attribute__((address_space(1)))
#define AS3 __attribute__((address_space(3)))

#if defined(__has_builtin)
#  if __has_builtin(__builtin_amdgcn_global_load_async_to_lds_b128)
#    define HAS_ASYNC_LDS 1
#  else
#    define HAS_ASYNC_LDS 0
#  endif
#  if __has_builtin(__builtin_amdgcn_s_wait_asynccnt)
#    define HAS_WAIT_ASYNC 1
#  else
#    define HAS_WAIT_ASYNC 0
#  endif
#else
#  define HAS_ASYNC_LDS 0
#  define HAS_WAIT_ASYNC 0
#endif

// ---------------- degree / normalization ----------------

__global__ void k_fill1(float* __restrict__ deg, int n) {
    int i = blockIdx.x * 256 + threadIdx.x;
    if (i < n) deg[i] = 1.0f;               // self-loop contributes 1 to every node
}

__global__ void k_count(const long long* __restrict__ ei, float* __restrict__ deg, int E) {
    int e = blockIdx.x * 256 + threadIdx.x;
    if (e < E) atomicAdd(&deg[ei[(size_t)E + e]], 1.0f);   // dst row of edge_index
}

__global__ void k_rsqrt(float* __restrict__ d, int n) {
    int i = blockIdx.x * 256 + threadIdx.x;
    if (i < n) d[i] = rsqrtf(d[i]);         // deg >= 1 always (self loops)
}

// ---------------- WMMA GEMM: D[rows x 64] = A[rows x K] * B[K x 64] ----------------
// B columns 0..31 come from Bl (row stride ldb), columns 32..63 from Br (row stride ldb).
// GEMM1: Bl=W1, Br=W1+32, ldb=64.  GEMM2: Bl=Wmu, Br=Wls, ldb=32.
// rows % 32 == 0, K % 32 == 0. 256 threads = 8 waves; each wave does one 16x16 tile.
__global__ void k_gemm_wmma(const float* __restrict__ A,
                            const float* __restrict__ Bl,
                            const float* __restrict__ Br,
                            float* __restrict__ D,
                            int lda, int ldb, int K) {
    // A tile staged as f32 (async DMA target), converted to bf16 at fragment build.
    __shared__ __attribute__((aligned(16))) float  As32[32][36]; // pitch 144B (mult of 16)
    __shared__ __attribute__((aligned(16))) __bf16 BsT[64][40];  // transposed [col][k], pitch 80B

    const int tid  = threadIdx.x;
    const int lane = tid & 31;
    const int wave = tid >> 5;
    const int half = lane >> 4;     // 0: lanes 0-15, 1: lanes 16-31
    const int m    = lane & 15;
    const int tr   = wave >> 2;     // tile row within block (0..1)
    const int tc   = wave & 3;      // tile col within block (0..3)
    const int row0 = blockIdx.x * 32;

    // Per-thread staging assignments (computed once; branchless inner loop).
    const int rA  = tid >> 3, c4A = tid & 7;                 // A: one float4 per thread
    const float* gA = A + (size_t)(row0 + rA) * lda + c4A * 4;
    const int cB  = tid & 63;                                // B: column owned by this thread
    const int g0  = tid >> 6;                                // k-group 0..3 (and +4)
    const float* pB = (cB < 32) ? (Bl + cB) : (Br + (cB - 32));

    v8f acc = {};

    for (int k0 = 0; k0 < K; k0 += 32) {
        // ---- stage A tile 32x32 f32 (async global->LDS if available) ----
#if HAS_ASYNC_LDS
        __builtin_amdgcn_global_load_async_to_lds_b128(
            (AS1 i32x4*)const_cast<float*>(gA + k0),
            (AS3 i32x4*)&As32[rA][c4A * 4], 0, 0);
#else
        *(float4*)&As32[rA][c4A * 4] = *(const float4*)(gA + k0);
#endif
        // ---- stage B tile transposed: BsT[col][k], 4 k's per thread x 2 groups ----
        #pragma unroll
        for (int w = 0; w < 2; ++w) {
            const int g  = g0 + w * 4;
            const int kb = k0 + g * 4;
            const float f0 = pB[(size_t)(kb + 0) * ldb];
            const float f1 = pB[(size_t)(kb + 1) * ldb];
            const float f2 = pB[(size_t)(kb + 2) * ldb];
            const float f3 = pB[(size_t)(kb + 3) * ldb];
            BsT[cB][g * 4 + 0] = (__bf16)f0;   // contiguous 8B run -> ds_store_b64
            BsT[cB][g * 4 + 1] = (__bf16)f1;
            BsT[cB][g * 4 + 2] = (__bf16)f2;
            BsT[cB][g * 4 + 3] = (__bf16)f3;
        }
#if HAS_ASYNC_LDS
#  if HAS_WAIT_ASYNC
        __builtin_amdgcn_s_wait_asynccnt(0);
#  else
        asm volatile("s_wait_asynccnt 0" ::: "memory");
#  endif
#endif
        __syncthreads();

        // ---- build fragments (packed 16B LDS loads) ----
        // A (16-bit 16x32 layout): lane (m, half h) element i -> K = (i<8 ? i : i+8) + 8h.
        // Elements 0..7 = f32 K [8h, 8h+8), elements 8..15 = f32 K [16+8h, 24+8h).
        const int rowa = tr * 16 + m;
        const float4 f0 = *(const float4*)&As32[rowa][8 * half + 0];
        const float4 f1 = *(const float4*)&As32[rowa][8 * half + 4];
        const float4 f2 = *(const float4*)&As32[rowa][16 + 8 * half + 0];
        const float4 f3 = *(const float4*)&As32[rowa][16 + 8 * half + 4];
        v16bf a;
        a[0]  = (__bf16)f0.x; a[1]  = (__bf16)f0.y; a[2]  = (__bf16)f0.z; a[3]  = (__bf16)f0.w;
        a[4]  = (__bf16)f1.x; a[5]  = (__bf16)f1.y; a[6]  = (__bf16)f1.z; a[7]  = (__bf16)f1.w;
        a[8]  = (__bf16)f2.x; a[9]  = (__bf16)f2.y; a[10] = (__bf16)f2.z; a[11] = (__bf16)f2.w;
        a[12] = (__bf16)f3.x; a[13] = (__bf16)f3.y; a[14] = (__bf16)f3.z; a[15] = (__bf16)f3.w;

        // B (16-bit 32x16 layout): lane element i -> K = (i<8 ? i : i+8) + 8h, col N = lane&15.
        // With BsT[col][k] this is two runs of 8 contiguous bf16 -> two ds_load_b128.
        const int colb = tc * 16 + m;
        const v8bf b_lo = *(const v8bf*)&BsT[colb][8 * half + 0];
        const v8bf b_hi = *(const v8bf*)&BsT[colb][16 + 8 * half];
        const v16bf b = __builtin_shufflevector(b_lo, b_hi,
            0, 1, 2, 3, 4, 5, 6, 7, 8, 9, 10, 11, 12, 13, 14, 15);

        acc = __builtin_amdgcn_wmma_f32_16x16x32_bf16(
            /*neg_a=*/false, a, /*neg_b=*/false, b,
            /*c_mod=*/(short)0, acc, /*reuse_a=*/false, /*reuse_b=*/false);
        __syncthreads();
    }

    // D layout: VGPR r -> M = r + 8*half, N = lane&15.
    #pragma unroll
    for (int r = 0; r < 8; ++r) {
        D[(size_t)(row0 + tr * 16 + r + 8 * half) * 64 + tc * 16 + m] = acc[r];
    }
}

// ---------------- aggregation (D^-1/2 A D^-1/2 message passing) ----------------

// Self-loop term initializes the accumulator: out[i,:] = h[i,:] * dinv[i]^2
__global__ void k_selfinit(const float* __restrict__ h, const float* __restrict__ dinv,
                           float* __restrict__ out, int total) {
    int idx = blockIdx.x * 256 + threadIdx.x;
    if (idx < total) {
        float di = dinv[idx >> 6];
        out[idx] = h[idx] * di * di;
    }
}

// One 16-lane group per edge; each lane handles 4 channels (float4 coalesced gather,
// 4x GLOBAL_ATOMIC_ADD_F32 scatter).
__global__ void k_scatter(const long long* __restrict__ ei, const float* __restrict__ dinv,
                          const float* __restrict__ h, float* __restrict__ out, int E) {
    long long idx = (long long)blockIdx.x * 256 + threadIdx.x;
    long long e = idx >> 4;
    int t = (int)(idx & 15);
    if (e >= E) return;
    long long s = ei[e];
    long long d = ei[(long long)E + e];
    float w = dinv[s] * dinv[d];
    float4 v = *(const float4*)(h + s * 64 + t * 4);
    float* o = out + d * 64 + t * 4;
    atomicAdd(o + 0, v.x * w);
    atomicAdd(o + 1, v.y * w);
    atomicAdd(o + 2, v.z * w);
    atomicAdd(o + 3, v.w * w);
}

__global__ void k_bias_relu(const float* __restrict__ agg, const float* __restrict__ b,
                            float* __restrict__ h, int total) {
    int idx = blockIdx.x * 256 + threadIdx.x;
    if (idx < total) h[idx] = fmaxf(agg[idx] + b[idx & 63], 0.0f);
}

// Split fused [mu|logstd] accumulator into output tuple layout (mu flat, then logstd flat).
__global__ void k_final(const float* __restrict__ agg, const float* __restrict__ bmu,
                        const float* __restrict__ bls, float* __restrict__ out, int n) {
    int idx = blockIdx.x * 256 + threadIdx.x;
    if (idx >= n * 64) return;
    int i = idx >> 6, c = idx & 63;
    if (c < 32) out[(size_t)i * 32 + c] = agg[idx] + bmu[c];
    else        out[(size_t)n * 32 + (size_t)i * 32 + (c - 32)] = agg[idx] + bls[c - 32];
}

// ---------------- host orchestration ----------------

extern "C" void kernel_launch(void* const* d_in, const int* in_sizes, int n_in,
                              void* d_out, int out_size, void* d_ws, size_t ws_size,
                              hipStream_t stream) {
    const float*     x   = (const float*)d_in[0];
    const long long* ei  = (const long long*)d_in[1];
    const float*     W1  = (const float*)d_in[2];
    const float*     b1  = (const float*)d_in[3];
    const float*     Wmu = (const float*)d_in[4];
    const float*     bmu = (const float*)d_in[5];
    const float*     Wls = (const float*)d_in[6];
    const float*     bls = (const float*)d_in[7];
    float* out = (float*)d_out;

    const int N = in_sizes[0] / IN_C;   // 100000 (multiple of 32)
    const int E = in_sizes[1] / 2;      // 1600000

    // Workspace: dinv[N] | bufA[N*64] | bufB[N*64] | bufC[N*64]  (~77 MB)
    float* dinv = (float*)d_ws;
    float* bufA = dinv + (size_t)N;            // h0, then h (after relu)
    float* bufB = bufA + (size_t)N * 64;       // agg1, then z = [h*Wmu | h*Wls]
    float* bufC = bufB + (size_t)N * 64;       // agg2

    auto cdiv = [](long long a, long long b) { return (int)((a + b - 1) / b); };
    const int total = N * 64;

    // Normalization (computed once, shared by all three GCN convs).
    k_fill1<<<cdiv(N, 256), 256, 0, stream>>>(dinv, N);
    k_count<<<cdiv(E, 256), 256, 0, stream>>>(ei, dinv, E);
    k_rsqrt<<<cdiv(N, 256), 256, 0, stream>>>(dinv, N);

    // Layer 1: h0 = x @ W1 (WMMA bf16), aggregate, bias+relu.
    k_gemm_wmma<<<N / 32, 256, 0, stream>>>(x, W1, W1 + 32, bufA, IN_C, HID, IN_C);
    k_selfinit<<<cdiv(total, 256), 256, 0, stream>>>(bufA, dinv, bufB, total);
    k_scatter<<<cdiv((long long)E * 16, 256), 256, 0, stream>>>(ei, dinv, bufA, bufB, E);
    k_bias_relu<<<cdiv(total, 256), 256, 0, stream>>>(bufB, b1, bufA, total);

    // Layer 2 (fused mu+logstd): z = h @ [Wmu|Wls], single aggregation pass.
    k_gemm_wmma<<<N / 32, 256, 0, stream>>>(bufA, Wmu, Wls, bufB, HID, 32, HID);
    k_selfinit<<<cdiv(total, 256), 256, 0, stream>>>(bufB, dinv, bufC, total);
    k_scatter<<<cdiv((long long)E * 16, 256), 256, 0, stream>>>(ei, dinv, bufB, bufC, E);
    k_final<<<cdiv(total, 256), 256, 0, stream>>>(bufC, bmu, bls, out, N);
}
